// VQ_VAE_78451872629108
// MI455X (gfx1250) — matrix operand order repaired and checked by reading, not compiled, not run on hardware
//
#include <hip/hip_runtime.h>
#include <hip/hip_bf16.h>

// ---------------------------------------------------------------------------
// CDNA5 (gfx1250) VQ-VAE forward.  All conv / tconv / VQ-distance GEMMs run on
// v_wmma_f32_16x16x32_f16 (wave32).  NHWC f16 activations, pre-packed f16
// weights in B-fragment order, f32 accumulation.
// Conv macro-tile: 32 spatial rows x 32 out-channels per wave (4 WMMAs/chunk).
// ---------------------------------------------------------------------------

typedef __attribute__((ext_vector_type(16))) _Float16 v16h;
typedef __attribute__((ext_vector_type(8)))  _Float16 v8h;
typedef __attribute__((ext_vector_type(8)))  float    v8f;

__device__ __forceinline__ v8f wmma_f16(v16h a, v16h b, v8f c) {
    return __builtin_amdgcn_wmma_f32_16x16x32_f16(
        /*neg_a=*/false, a, /*neg_b=*/false, b,
        /*c_mod=*/(short)0, c, /*reuse_a=*/false, /*reuse_b=*/false);
}

// ---------------------------------------------------------------------------
// Implicit-GEMM conv over NHWC f16 activations.
//   MODE==0 : normal conv, taps t = ky*KW+kx, K index k = t*Cin + ci
//   MODE==1 : ConvTranspose2d k=4,s=2,p=1; blockIdx.z = output parity class,
//             taps t in [0,4): ky = kyp+2*(t>>1), kx = kxp+2*(t&1)
// Weights pre-packed: wpack[n * Ktotp + k] (f16, zero padded to Npad x Ktotp;
// MODE==1 has 4 consecutive parity blocks).
// Each wave: 32 (spatial rows, two 16-row fragments) x 32 (out-channels, two
// 16-wide fragments) -> 4 accumulators, 4 WMMAs per 32-K chunk.
// Fast path (Cin%32==0): A fragment = two 16B loads; B = one 32B load/tile.
// ---------------------------------------------------------------------------
template <int MODE>
__global__ void __launch_bounds__(256)
conv_wmma_kernel(const _Float16* __restrict__ act,
                 const _Float16* __restrict__ wpack,
                 const float*    __restrict__ bias,
                 const _Float16* __restrict__ resid,   // optional residual (NHWC)
                 _Float16*       __restrict__ out16,   // optional f16 out (NHWC)
                 float*          __restrict__ out32,   // optional f32 out
                 int out32_nchw,
                 int Bn, int Cin, int Hin, int Win,
                 int Cout, int Hout, int Wout,
                 int KH, int KW, int stride, int pad,
                 int relu_in, int relu_out,
                 int Ktotp, int Npad)
{
    const int lane = threadIdx.x & 31;
    const int wave = threadIdx.x >> 5;
    const int g    = lane >> 4;     // wave half (0/1)
    const int ln   = lane & 15;

    int Hm = Hout, Wm = Wout, py = 0, px = 0;
    if (MODE == 1) { Hm = Hout >> 1; Wm = Wout >> 1; py = blockIdx.z >> 1; px = blockIdx.z & 1; }
    const int ntaps = (MODE == 1) ? 4 : KH * KW;
    const int Ktot  = Cin * ntaps;
    const int HWm   = Hm * Wm;
    const int Mtot  = Bn * HWm;
    const int mtile = blockIdx.x * 8 + wave;     // 32 rows per tile
    if (mtile * 32 >= Mtot) return;              // uniform per wave

    const int kyp = (MODE == 1) ? ((py + 1) & 1) : 0;
    const int kxp = (MODE == 1) ? ((px + 1) & 1) : 0;

    // ---- two A row-fragments owned by this lane (row = ln within fragment)
    int  bA[2], oyA[2], oxA[2];
    bool vAr[2];
#pragma unroll
    for (int ms = 0; ms < 2; ++ms) {
        const int mA = mtile * 32 + ms * 16 + ln;
        vAr[ms] = (mA < Mtot);
        const int mc = vAr[ms] ? mA : 0;
        bA[ms] = mc / HWm;
        const int rA = mc - bA[ms] * HWm;
        int oy = rA / Wm;
        int ox = rA - oy * Wm;
        if (MODE == 1) { oy = 2 * oy + py; ox = 2 * ox + px; }
        oyA[ms] = oy; oxA[ms] = ox;
    }

    // ---- two B row-fragments (two 16-wide column tiles per wave)
    const int n0 = blockIdx.y * 32 + ln;
    const int n1 = n0 + 16;
    const _Float16* wp = wpack;
    if (MODE == 1) wp += (size_t)blockIdx.z * Npad * Ktotp;
    const _Float16* wrow0 = wp + (size_t)n0 * Ktotp;
    const _Float16* wrow1 = wp + (size_t)n1 * Ktotp;

    v8f acc00 = {}, acc01 = {}, acc10 = {}, acc11 = {};

    if ((Cin & 31) == 0) {
        // ================= fast path: vector loads =================
        for (int t = 0; t < ntaps; ++t) {
            const _Float16* ap[2];
            bool inb[2];
#pragma unroll
            for (int ms = 0; ms < 2; ++ms) {
                int iy, ix;
                if (MODE == 1) {
                    int ky = kyp + ((t >> 1) << 1);
                    int kx = kxp + ((t & 1) << 1);
                    iy = (oyA[ms] + 1 - ky) >> 1;
                    ix = (oxA[ms] + 1 - kx) >> 1;
                } else {
                    int ky = t / KW;
                    int kx = t - ky * KW;
                    iy = oyA[ms] * stride - pad + ky;
                    ix = oxA[ms] * stride - pad + kx;
                }
                inb[ms] = vAr[ms] && iy >= 0 && iy < Hin && ix >= 0 && ix < Win;
                ap[ms]  = act + ((size_t)(bA[ms] * Hin + iy) * Win + ix) * Cin;
            }

            for (int cb = 0; cb < Cin; cb += 32) {
                const int k0 = t * Cin + cb;
                v16h Af[2];
#pragma unroll
                for (int ms = 0; ms < 2; ++ms) {
                    v8h s0 = {}, s1 = {};
                    if (inb[ms]) {
                        s0 = *(const v8h*)(ap[ms] + cb + g * 16);
                        s1 = *(const v8h*)(ap[ms] + cb + g * 16 + 8);
                    }
                    if (relu_in) {
#pragma unroll
                        for (int e = 0; e < 8; ++e) {
                            s0[e] = (s0[e] > (_Float16)0.f) ? s0[e] : (_Float16)0.f;
                            s1[e] = (s1[e] > (_Float16)0.f) ? s1[e] : (_Float16)0.f;
                        }
                    }
#pragma unroll
                    for (int e = 0; e < 8; ++e) { Af[ms][e] = s0[e]; Af[ms][e + 8] = s1[e]; }
                }
                v16h Bf0 = *(const v16h*)(wrow0 + k0 + g * 16);
                v16h Bf1 = *(const v16h*)(wrow1 + k0 + g * 16);
                acc00 = wmma_f16(Af[0], Bf0, acc00);
                acc01 = wmma_f16(Af[0], Bf1, acc01);
                acc10 = wmma_f16(Af[1], Bf0, acc10);
                acc11 = wmma_f16(Af[1], Bf1, acc11);
            }
        }
    } else {
        // ================= slow path (Cin=3 input conv) =================
        for (int k0 = 0; k0 < Ktotp; k0 += 32) {
            v16h Af[2];
#pragma unroll
            for (int ms = 0; ms < 2; ++ms) {
#pragma unroll
                for (int e = 0; e < 16; ++e) {
                    int ka = k0 + g * 16 + e;
                    float v = 0.f;
                    if (vAr[ms] && ka < Ktot) {
                        int t  = ka / Cin;
                        int ci = ka - t * Cin;
                        int iy, ix;
                        if (MODE == 1) {
                            int ky = kyp + ((t >> 1) << 1);
                            int kx = kxp + ((t & 1) << 1);
                            iy = (oyA[ms] + 1 - ky) >> 1;
                            ix = (oxA[ms] + 1 - kx) >> 1;
                        } else {
                            int ky = t / KW;
                            int kx = t - ky * KW;
                            iy = oyA[ms] * stride - pad + ky;
                            ix = oxA[ms] * stride - pad + kx;
                        }
                        if (iy >= 0 && iy < Hin && ix >= 0 && ix < Win) {
                            v = (float)act[((size_t)(bA[ms] * Hin + iy) * Win + ix) * Cin + ci];
                            if (relu_in) v = fmaxf(v, 0.f);
                        }
                    }
                    Af[ms][e] = (_Float16)v;
                }
            }
            v16h Bf0 = *(const v16h*)(wrow0 + k0 + g * 16);
            v16h Bf1 = *(const v16h*)(wrow1 + k0 + g * 16);
            acc00 = wmma_f16(Af[0], Bf0, acc00);
            acc01 = wmma_f16(Af[0], Bf1, acc01);
            acc10 = wmma_f16(Af[1], Bf0, acc10);
            acc11 = wmma_f16(Af[1], Bf1, acc11);
        }
    }

    // ---- store (C/D f32 layout: col = ln, rows = g*8 + r)
    const float bv0 = (n0 < Cout && bias) ? bias[n0] : 0.f;
    const float bv1 = (n1 < Cout && bias) ? bias[n1] : 0.f;
#pragma unroll
    for (int ms = 0; ms < 2; ++ms) {
        const v8f& a0 = ms ? acc10 : acc00;
        const v8f& a1 = ms ? acc11 : acc01;
#pragma unroll
        for (int r = 0; r < 8; ++r) {
            int mr = mtile * 32 + ms * 16 + g * 8 + r;
            if (mr < Mtot) {
                int bb = mr / HWm;
                int rr = mr - bb * HWm;
                int oy = rr / Wm;
                int ox = rr - oy * Wm;
                if (MODE == 1) { oy = 2 * oy + py; ox = 2 * ox + px; }
                const size_t sp = ((size_t)(bb * Hout + oy) * Wout + ox);
                if (n0 < Cout) {
                    size_t i16 = sp * Cout + n0;
                    float v = a0[r] + bv0;
                    if (resid) v += (float)resid[i16];
                    if (relu_out) v = fmaxf(v, 0.f);
                    if (out16) out16[i16] = (_Float16)v;
                    if (out32) out32[out32_nchw ? (((size_t)(bb * Cout + n0) * Hout + oy) * Wout + ox) : i16] = v;
                }
                if (n1 < Cout) {
                    size_t i16 = sp * Cout + n1;
                    float v = a1[r] + bv1;
                    if (resid) v += (float)resid[i16];
                    if (relu_out) v = fmaxf(v, 0.f);
                    if (out16) out16[i16] = (_Float16)v;
                    if (out32) out32[out32_nchw ? (((size_t)(bb * Cout + n1) * Hout + oy) * Wout + ox) : i16] = v;
                }
            }
        }
    }
}

// ---------------------------------------------------------------------------
// Weight packing:  wpack[n * Ktotp + k], k = t*Cin + ci  (f16, zero padded)
// ---------------------------------------------------------------------------
__global__ void pack_conv_w_kernel(const float* __restrict__ w, _Float16* __restrict__ wp,
                                   int Cout, int Cin, int KH, int KW,
                                   int Ktot, int Ktotp, int Npad)
{
    int i = blockIdx.x * blockDim.x + threadIdx.x;
    if (i >= Npad * Ktotp) return;
    int n = i / Ktotp;
    int k = i - n * Ktotp;
    float v = 0.f;
    if (n < Cout && k < Ktot) {
        int t  = k / Cin;
        int ci = k - t * Cin;
        int ky = t / KW;
        int kx = t - ky * KW;
        v = w[((size_t)(n * Cin + ci) * KH + ky) * KW + kx];   // OIHW
    }
    wp[i] = (_Float16)v;
}

__global__ void pack_tconv_w_kernel(const float* __restrict__ w, _Float16* __restrict__ wp,
                                    int Cout, int Cin, int Ktot, int Ktotp, int Npad)
{
    int i = blockIdx.x * blockDim.x + threadIdx.x;
    const int per = Npad * Ktotp;
    if (i >= 4 * per) return;
    int p = i / per;
    int r = i - p * per;
    int n = r / Ktotp;
    int k = r - n * Ktotp;
    const int kyp = ((p >> 1) + 1) & 1;
    const int kxp = ((p & 1) + 1) & 1;
    float v = 0.f;
    if (n < Cout && k < Ktot) {
        int t  = k / Cin;
        int ci = k - t * Cin;
        int ky = kyp + ((t >> 1) << 1);
        int kx = kxp + ((t & 1) << 1);
        v = w[((size_t)(ci * Cout + n) * 4 + ky) * 4 + kx];    // IOHW
    }
    wp[i] = (_Float16)v;
}

// ---------------------------------------------------------------------------
// VQ argmin: score = ||c||^2 - 2 z.c via WMMA.  z row-major [N,64] f16.
// ---------------------------------------------------------------------------
__global__ void __launch_bounds__(256)
vq_argmin_kernel(const _Float16* __restrict__ z16,   // [N,64]
                 const _Float16* __restrict__ cb16,  // [K,64]
                 const float*    __restrict__ cnorm, // [K]
                 int* __restrict__ idxOut, int N)
{
    const int lane = threadIdx.x & 31;
    const int wave = threadIdx.x >> 5;
    const int g    = lane >> 4;
    const int ln   = lane & 15;
    const int mtile = blockIdx.x * 8 + wave;
    if (mtile * 16 >= N) return;

    const _Float16* zr = z16 + (size_t)(mtile * 16 + ln) * 64;
    v8h a0 = *(const v8h*)(zr + g * 16);
    v8h a1 = *(const v8h*)(zr + 8 + g * 16);
    v8h a2 = *(const v8h*)(zr + 32 + g * 16);
    v8h a3 = *(const v8h*)(zr + 40 + g * 16);
    v16h A0, A1;
#pragma unroll
    for (int e = 0; e < 8; ++e) {
        A0[e] = a0[e]; A0[e + 8] = a1[e];
        A1[e] = a2[e]; A1[e + 8] = a3[e];
    }

    float minv[8];
    int   mini[8];
#pragma unroll
    for (int r = 0; r < 8; ++r) { minv[r] = 3.4e38f; mini[r] = 0; }

    for (int ct = 0; ct < 64; ++ct) {       // K = 1024 codes / 16
        const int c = ct * 16 + ln;
        const _Float16* cr = cb16 + (size_t)c * 64;
        v16h B0 = *(const v16h*)(cr + g * 16);
        v16h B1 = *(const v16h*)(cr + 32 + g * 16);
        v8f acc = {};
        acc = wmma_f16(A0, B0, acc);
        acc = wmma_f16(A1, B1, acc);
        const float cn = cnorm[c];
#pragma unroll
        for (int r = 0; r < 8; ++r) {
            float s = cn - 2.f * acc[r];
            if (s < minv[r]) { minv[r] = s; mini[r] = c; }
        }
    }

    // reduce across the 16 lanes of each half (same rows, different codes)
#pragma unroll
    for (int off = 8; off >= 1; off >>= 1) {
#pragma unroll
        for (int r = 0; r < 8; ++r) {
            float ov = __shfl_xor(minv[r], off, 32);
            int   oi = __shfl_xor(mini[r], off, 32);
            if (ov < minv[r] || (ov == minv[r] && oi < mini[r])) { minv[r] = ov; mini[r] = oi; }
        }
    }
    if (ln == 0) {
#pragma unroll
        for (int r = 0; r < 8; ++r)
            idxOut[mtile * 16 + g * 8 + r] = mini[r];
    }
}

// ---------------------------------------------------------------------------
// Small helper kernels
// ---------------------------------------------------------------------------
__global__ void nchw_to_nhwc_f16_kernel(const float* __restrict__ in,
                                        _Float16* __restrict__ out,
                                        int C, int H, int W, int total) {
    int i = blockIdx.x * blockDim.x + threadIdx.x;
    if (i >= total) return;
    int c = i % C;
    int p = i / C;
    int x = p % W;  p /= W;
    int y = p % H;
    int b = p / H;
    out[i] = (_Float16)in[((size_t)(b * C + c) * H + y) * W + x];
}

__global__ void vq_prep_kernel(const float* __restrict__ cb,
                               _Float16* __restrict__ cb16,
                               float* __restrict__ cnorm) {
    const int k = blockIdx.x, d = threadIdx.x;   // 64 threads
    float v = cb[k * 64 + d];
    cb16[k * 64 + d] = (_Float16)v;
    __shared__ float red[64];
    red[d] = v * v;
    __syncthreads();
    for (int s = 32; s > 0; s >>= 1) { if (d < s) red[d] += red[d + s]; __syncthreads(); }
    if (d == 0) cnorm[k] = red[0];
}

__global__ void zero_kernel(float* p, int n) {
    int i = blockIdx.x * blockDim.x + threadIdx.x;
    if (i < n) p[i] = 0.f;
}

__global__ void vq_count_kernel(const int* __restrict__ idx, float* counts, int N) {
    int i = blockIdx.x * blockDim.x + threadIdx.x;
    if (i < N) atomicAdd(&counts[idx[i]], 1.0f);  // exact integer sums -> deterministic
}

__global__ void __launch_bounds__(256)
vq_stats_kernel(const float* __restrict__ z32,   // [N,64]
                const float* __restrict__ cb,    // [K,64]
                const int* __restrict__ idx, _Float16* __restrict__ q16,  // [N,64]
                float* __restrict__ partial, int N, int D)
{
    const int tid = blockIdx.x * 256 + threadIdx.x;
    float local = 0.f;
    if (tid < N * D) {
        int row = tid / D;
        int d   = tid - row * D;
        float zv = z32[tid];
        float qv = cb[(size_t)idx[row] * D + d];
        float df = qv - zv;
        local = df * df;
        q16[tid] = (_Float16)qv;   // straight-through forward value == q
    }
    __shared__ float red[256];
    red[threadIdx.x] = local;
    __syncthreads();
    for (int s = 128; s > 0; s >>= 1) {
        if (threadIdx.x < s) red[threadIdx.x] += red[threadIdx.x + s];
        __syncthreads();
    }
    if (threadIdx.x == 0) partial[blockIdx.x] = red[0];
}

__global__ void __launch_bounds__(1024)
vq_final_kernel(const float* __restrict__ partial, int nPartial,
                const float* __restrict__ counts, int K,
                float invND, float invN, float beta,
                float* __restrict__ outScalars)
{
    __shared__ float red[1024];
    float s = 0.f;
    for (int i = threadIdx.x; i < nPartial; i += 1024) s += partial[i];
    red[threadIdx.x] = s;
    __syncthreads();
    for (int st = 512; st > 0; st >>= 1) {
        if (threadIdx.x < st) red[threadIdx.x] += red[threadIdx.x + st];
        __syncthreads();
    }
    float sse = red[0];
    __syncthreads();
    float h = 0.f;
    for (int k = threadIdx.x; k < K; k += 1024) {
        float p = counts[k] * invN;
        h += p * logf(p + 1e-10f);
    }
    red[threadIdx.x] = h;
    __syncthreads();
    for (int st = 512; st > 0; st >>= 1) {
        if (threadIdx.x < st) red[threadIdx.x] += red[threadIdx.x + st];
        __syncthreads();
    }
    if (threadIdx.x == 0) {
        outScalars[0] = (1.f + beta) * (sse * invND);  // vq_loss = q_loss + beta*e_loss
        outScalars[1] = expf(-red[0]);                 // perplexity
    }
}

// ---------------------------------------------------------------------------
// Host-side launch helpers
// ---------------------------------------------------------------------------
static inline void run_conv(hipStream_t s, const _Float16* act, const float* w,
                            const float* bias, const _Float16* resid,
                            _Float16* o16, float* o32, int o32nchw,
                            _Float16* wpack,
                            int Bn, int Cin, int Hin, int Win,
                            int Cout, int Hout, int Wout,
                            int KH, int KW, int stride, int pad,
                            int relu_in, int relu_out)
{
    const int Ktot  = Cin * KH * KW;
    const int Ktotp = (Ktot + 31) & ~31;
    const int Npad  = (Cout + 31) & ~31;
    {
        int n = Npad * Ktotp;
        hipLaunchKernelGGL(pack_conv_w_kernel, dim3((n + 255) / 256), dim3(256), 0, s,
                           w, wpack, Cout, Cin, KH, KW, Ktot, Ktotp, Npad);
    }
    long long Mtot = (long long)Bn * Hout * Wout;
    int mt = (int)((Mtot + 31) / 32);                 // 32 rows per wave-tile
    dim3 grid((mt + 7) / 8, Npad / 32, 1);
    hipLaunchKernelGGL((conv_wmma_kernel<0>), grid, dim3(256), 0, s,
                       act, wpack, bias, resid, o16, o32, o32nchw,
                       Bn, Cin, Hin, Win, Cout, Hout, Wout,
                       KH, KW, stride, pad, relu_in, relu_out, Ktotp, Npad);
}

static inline void run_tconv(hipStream_t s, const _Float16* act, const float* w,
                             const float* bias, _Float16* o16, _Float16* wpack,
                             int Bn, int Cin, int Hin, int Win,
                             int Cout, int Hout, int Wout)
{
    const int Ktot  = Cin * 4;
    const int Ktotp = (Ktot + 31) & ~31;
    const int Npad  = (Cout + 31) & ~31;
    {
        int n = 4 * Npad * Ktotp;
        hipLaunchKernelGGL(pack_tconv_w_kernel, dim3((n + 255) / 256), dim3(256), 0, s,
                           w, wpack, Cout, Cin, Ktot, Ktotp, Npad);
    }
    long long Mtot = (long long)Bn * (Hout / 2) * (Wout / 2);
    int mt = (int)((Mtot + 31) / 32);                 // 32 rows per wave-tile
    dim3 grid((mt + 7) / 8, Npad / 32, 4);
    hipLaunchKernelGGL((conv_wmma_kernel<1>), grid, dim3(256), 0, s,
                       act, wpack, bias, (const _Float16*)nullptr, o16, (float*)nullptr, 0,
                       Bn, Cin, Hin, Win, Cout, Hout, Wout,
                       4, 4, 2, 1, 0, 1, Ktotp, Npad);
}

// ---------------------------------------------------------------------------
extern "C" void kernel_launch(void* const* d_in, const int* in_sizes, int n_in,
                              void* d_out, int out_size, void* d_ws, size_t ws_size,
                              hipStream_t stream)
{
    (void)in_sizes; (void)n_in; (void)out_size; (void)ws_size;

    const float* x        = (const float*)d_in[0];
    const float* enc_w0   = (const float*)d_in[1];  const float* enc_b0   = (const float*)d_in[2];
    const float* enc_w1   = (const float*)d_in[3];  const float* enc_b1   = (const float*)d_in[4];
    const float* enc_w2   = (const float*)d_in[5];  const float* enc_b2   = (const float*)d_in[6];
    const float* enc_r0_w1= (const float*)d_in[7];  const float* enc_r0_b1= (const float*)d_in[8];
    const float* enc_r0_w2= (const float*)d_in[9];  const float* enc_r0_b2= (const float*)d_in[10];
    const float* enc_r1_w1= (const float*)d_in[11]; const float* enc_r1_b1= (const float*)d_in[12];
    const float* enc_r1_w2= (const float*)d_in[13]; const float* enc_r1_b2= (const float*)d_in[14];
    const float* enc_adj_w= (const float*)d_in[15]; const float* enc_adj_b= (const float*)d_in[16];
    const float* codebook = (const float*)d_in[17];
    const float* dec_adj_w= (const float*)d_in[18]; const float* dec_adj_b= (const float*)d_in[19];
    const float* dec_r0_w1= (const float*)d_in[20]; const float* dec_r0_b1= (const float*)d_in[21];
    const float* dec_r0_w2= (const float*)d_in[22]; const float* dec_r0_b2= (const float*)d_in[23];
    const float* dec_r1_w1= (const float*)d_in[24]; const float* dec_r1_b1= (const float*)d_in[25];
    const float* dec_r1_w2= (const float*)d_in[26]; const float* dec_r1_b2= (const float*)d_in[27];
    const float* dec_tw0  = (const float*)d_in[28]; const float* dec_tb0  = (const float*)d_in[29];
    const float* dec_tw1  = (const float*)d_in[30]; const float* dec_tb1  = (const float*)d_in[31];
    const float* dec_tw2  = (const float*)d_in[32]; const float* dec_tb2  = (const float*)d_in[33];
    const float* out_w    = (const float*)d_in[34]; const float* out_b    = (const float*)d_in[35];

    // ---- workspace layout (256B aligned regions)
    char* ws = (char*)d_ws;
    size_t off = 0;
    auto alloc = [&](size_t bytes) -> char* {
        char* p = ws + off;
        off += (bytes + 255) & ~(size_t)255;
        return p;
    };
    _Float16* buf0   = (_Float16*)alloc((size_t)32 * 64 * 128 * 128 * 2);  //  64 MB
    _Float16* buf1   = (_Float16*)alloc((size_t)32 * 32 * 256 * 256 * 2);  // 128 MB
    _Float16* buf2   = (_Float16*)alloc((size_t)32 * 64 * 32 * 32 * 2);
    _Float16* z16    = (_Float16*)alloc((size_t)32 * 64 * 32 * 32 * 2);    // [N,64]
    float*    z32    = (float*)   alloc((size_t)32 * 64 * 32 * 32 * 4);    // [N,64]
    _Float16* q16    = (_Float16*)alloc((size_t)32 * 64 * 32 * 32 * 2);    // [N,64]
    _Float16* cb16   = (_Float16*)alloc((size_t)1024 * 64 * 2);
    float*    cnorm  = (float*)   alloc((size_t)1024 * 4);
    int*      idx    = (int*)     alloc((size_t)32768 * 4);
    float*    counts = (float*)   alloc((size_t)1024 * 4);
    float*    partial= (float*)   alloc((size_t)8192 * 4);
    _Float16* wpack  = (_Float16*)alloc((size_t)1048576 * 2);              //   2 MB
    _Float16* xh     = buf1;  // NHWC f16 input staged at head of buf1 (dead after conv0)

    // ---- input NCHW f32 -> NHWC f16
    {
        int n = 32 * 3 * 256 * 256;
        hipLaunchKernelGGL(nchw_to_nhwc_f16_kernel, dim3((n + 255) / 256), dim3(256), 0, stream,
                           x, xh, 3, 256, 256, n);
    }

    // ---- encoder
    run_conv(stream, xh,   enc_w0, enc_b0, nullptr, buf0, nullptr, 0, wpack,
             32, 3, 256, 256,   64, 128, 128,  4, 4, 2, 1, 0, 1);
    run_conv(stream, buf0, enc_w1, enc_b1, nullptr, buf1, nullptr, 0, wpack,
             32, 64, 128, 128,  128, 64, 64,   4, 4, 2, 1, 0, 1);
    run_conv(stream, buf1, enc_w2, enc_b2, nullptr, buf0, nullptr, 0, wpack,
             32, 128, 64, 64,   256, 32, 32,   4, 4, 2, 1, 0, 1);
    // res block 0: x + conv1x1(relu(conv3x3(relu(x))))
    run_conv(stream, buf0, enc_r0_w1, enc_r0_b1, nullptr, buf2, nullptr, 0, wpack,
             32, 256, 32, 32,   64, 32, 32,    3, 3, 1, 1, 1, 0);
    run_conv(stream, buf2, enc_r0_w2, enc_r0_b2, buf0, buf1, nullptr, 0, wpack,
             32, 64, 32, 32,    256, 32, 32,   1, 1, 1, 0, 1, 0);
    // res block 1
    run_conv(stream, buf1, enc_r1_w1, enc_r1_b1, nullptr, buf2, nullptr, 0, wpack,
             32, 256, 32, 32,   64, 32, 32,    3, 3, 1, 1, 1, 0);
    run_conv(stream, buf2, enc_r1_w2, enc_r1_b2, buf1, buf0, nullptr, 0, wpack,
             32, 64, 32, 32,    256, 32, 32,   1, 1, 1, 0, 1, 0);
    // enc_adj 1x1 -> zlat (NHWC == row-major [N,64]; f16 for VQ GEMM, f32 for loss)
    run_conv(stream, buf0, enc_adj_w, enc_adj_b, nullptr, z16, z32, 0, wpack,
             32, 256, 32, 32,   64, 32, 32,    1, 1, 1, 0, 0, 0);

    // ---- vector quantization
    hipLaunchKernelGGL(vq_prep_kernel, dim3(1024), dim3(64), 0, stream, codebook, cb16, cnorm);
    hipLaunchKernelGGL(vq_argmin_kernel, dim3(256), dim3(256), 0, stream,
                       z16, cb16, cnorm, idx, 32768);
    hipLaunchKernelGGL(zero_kernel, dim3(4), dim3(256), 0, stream, counts, 1024);
    hipLaunchKernelGGL(vq_stats_kernel, dim3(8192), dim3(256), 0, stream,
                       z32, codebook, idx, q16, partial, 32768, 64);
    hipLaunchKernelGGL(vq_count_kernel, dim3(128), dim3(256), 0, stream, idx, counts, 32768);

    // ---- decoder
    run_conv(stream, q16,  dec_adj_w, dec_adj_b, nullptr, buf0, nullptr, 0, wpack,
             32, 64, 32, 32,    256, 32, 32,   1, 1, 1, 0, 0, 0);
    run_conv(stream, buf0, dec_r0_w1, dec_r0_b1, nullptr, buf2, nullptr, 0, wpack,
             32, 256, 32, 32,   64, 32, 32,    3, 3, 1, 1, 1, 0);
    run_conv(stream, buf2, dec_r0_w2, dec_r0_b2, buf0, buf1, nullptr, 0, wpack,
             32, 64, 32, 32,    256, 32, 32,   1, 1, 1, 0, 1, 0);
    run_conv(stream, buf1, dec_r1_w1, dec_r1_b1, nullptr, buf2, nullptr, 0, wpack,
             32, 256, 32, 32,   64, 32, 32,    3, 3, 1, 1, 1, 0);
    run_conv(stream, buf2, dec_r1_w2, dec_r1_b2, buf1, buf0, nullptr, 0, wpack,
             32, 64, 32, 32,    256, 32, 32,   1, 1, 1, 0, 1, 0);
    run_tconv(stream, buf0, dec_tw0, dec_tb0, buf1, wpack, 32, 256, 32, 32,   128, 64, 64);
    run_tconv(stream, buf1, dec_tw1, dec_tb1, buf0, wpack, 32, 128, 64, 64,   64, 128, 128);
    run_tconv(stream, buf0, dec_tw2, dec_tb2, buf1, wpack, 32, 64, 128, 128,  32, 256, 256);
    // out conv 1x1 -> f32 NCHW reconstruction directly into d_out
    run_conv(stream, buf1, out_w, out_b, nullptr, nullptr, (float*)d_out, 1, wpack,
             32, 32, 256, 256,  3, 256, 256,   1, 1, 1, 0, 0, 0);

    // ---- scalars: vq_loss, perplexity appended after x_recon
    hipLaunchKernelGGL(vq_final_kernel, dim3(1), dim3(1024), 0, stream,
                       partial, 8192, counts, 1024,
                       1.0f / (32768.0f * 64.0f), 1.0f / 32768.0f, 0.25f,
                       (float*)d_out + (size_t)32 * 3 * 256 * 256);
}
